// StandardLSTM_37958920962332
// MI455X (gfx1250) — compile-verified, matrix operand order checked by reading, not run on hardware
//
#include <hip/hip_runtime.h>
#include <cstdint>
#include <cstddef>

#define T_SEQ 188
#define BATCH 256
#define IN_DIM 128
#define HID 256
#define GATES (4 * HID)
#define KB_IH 4                 // K-blocks of 32 for input projection (K=128)
#define KB_HH 8                 // K-blocks of 32 for recurrence (K=256)
#define FRAG 512                // bf16 elems per fragment-group (32 lanes x 16)

typedef __attribute__((ext_vector_type(16))) __bf16 v16bf;
typedef __attribute__((ext_vector_type(8)))  __bf16 v8bf;
typedef __attribute__((ext_vector_type(8)))  float  v8f;

__device__ __forceinline__ float fast_sigmoid(float x) {
    return 1.0f / (1.0f + __expf(-x));
}
__device__ __forceinline__ float fast_tanh(float x) {
    const float t = __expf(2.0f * x);          // inf for large x -> NaN; only the -1 limit is correct
    return (t - 1.0f) / (t + 1.0f);
}

__device__ __forceinline__ v16bf frag_concat(v8bf lo, v8bf hi) {
    return __builtin_shufflevector(lo, hi, 0, 1, 2, 3, 4, 5, 6, 7,
                                           8, 9, 10, 11, 12, 13, 14, 15);
}
// A fragment from LDS: two 16B runs at +0 and +16 elems (CDNA5 16-bit A layout).
__device__ __forceinline__ v16bf lds_afrag(const __bf16* p) {
    return frag_concat(*(const v8bf*)p, *(const v8bf*)(p + 16));
}
// B fragment from pre-swizzled pack: uniform (SGPR) base + 32-bit element offset
// -> global_load_b128 saddr form; lane's 16 bf16 are contiguous (32B).
__device__ __forceinline__ v16bf pack_frag(const __bf16* __restrict__ base, int off) {
    return frag_concat(*(const v8bf*)(base + off), *(const v8bf*)(base + off + 8));
}

// One-shot prep: bf16-convert weights into per-lane WMMA fragment order, fold biases.
// Pack layout: [dir][tile n][kblock][lane][e0..e15]; element (n,kb,L,e) holds
// W[16n + (L&15)][32*kb + 8*(L>>4) + (e&7) + 16*(e>>3)].
__global__ void lstm_prep_kernel(const float* __restrict__ wih_f, const float* __restrict__ whh_f,
                                 const float* __restrict__ bih_f, const float* __restrict__ bhh_f,
                                 const float* __restrict__ wih_b, const float* __restrict__ whh_b,
                                 const float* __restrict__ bih_b, const float* __restrict__ bhh_b,
                                 __bf16* __restrict__ wih_pack, __bf16* __restrict__ whh_pack,
                                 float* __restrict__ bias) {
    const int i0 = blockIdx.x * blockDim.x + threadIdx.x;
    const int stride = gridDim.x * blockDim.x;

    for (int p = i0; p < 2 * 64 * KB_IH * FRAG; p += stride) {   // 262144
        const int d   = p >> 17;
        const int r   = p & 131071;
        const int n   = r >> 11;
        const int kbi = (r >> 9) & (KB_IH - 1);
        const int L   = (r >> 4) & 31;
        const int e   = r & 15;
        const int c   = 16 * n + (L & 15);
        const int k   = 32 * kbi + ((L >> 4) << 3) + (e & 7) + ((e >> 3) << 4);
        const float* src = d ? wih_b : wih_f;
        wih_pack[p] = (__bf16)src[c * IN_DIM + k];
    }
    for (int p = i0; p < 2 * 64 * KB_HH * FRAG; p += stride) {   // 524288
        const int d   = p >> 18;
        const int r   = p & 262143;
        const int n   = r >> 12;
        const int kbi = (r >> 9) & (KB_HH - 1);
        const int L   = (r >> 4) & 31;
        const int e   = r & 15;
        const int c   = 16 * n + (L & 15);
        const int k   = 32 * kbi + ((L >> 4) << 3) + (e & 7) + ((e >> 3) << 4);
        const float* src = d ? whh_b : whh_f;
        whh_pack[p] = (__bf16)src[c * HID + k];
    }
    for (int k = i0; k < GATES; k += stride) {
        bias[k]         = bih_f[k] + bhh_f[k];
        bias[GATES + k] = bih_b[k] + bhh_b[k];
    }
}

// Persistent batch-partitioned bidirectional LSTM scan.
// grid = 32 blocks: blockIdx>>4 = direction, blockIdx&15 = 16-row batch tile.
// 16 waves/block; wave w owns hidden slice [16w,16w+16) and its i,f,g,o gate tiles,
// so the cell update runs entirely in registers. Only workgroup barriers (2/step).
__global__ __launch_bounds__(512, 1)
void lstm_scan_kernel(const float* __restrict__ x,
                      const __bf16* __restrict__ wih_pack,
                      const __bf16* __restrict__ whh_pack,
                      const float* __restrict__ bias_all,
                      float* __restrict__ out) {
    // +8 bf16 row padding -> A-fragment ds_read_b128 hits distinct banks.
    __shared__ __align__(16) __bf16 h_lds[16][HID + 8];
    __shared__ __align__(16) __bf16 xa_lds[16][IN_DIM + 8];

    const int dir   = blockIdx.x >> 4;
    const int m0    = (blockIdx.x & 15) << 4;
    const int tid   = threadIdx.x;
    const int lane  = tid & 31;
    const int wv    = tid >> 5;
    const int l15   = lane & 15;
    const int khalf = (lane >> 4) << 3;
    const int mbase = (lane >> 4) << 3;

    // Block-uniform pack bases (stay in SGPRs -> saddr-form global loads).
    const __bf16* __restrict__ wbase = wih_pack + (size_t)dir * (64 * KB_IH * FRAG);
    const __bf16* __restrict__ hbase = whh_pack + (size_t)dir * (64 * KB_HH * FRAG);

    // Per-wave 32-bit element offsets into the packs (advance FRAG per K-block).
    int oih[4], ohh[4];
#pragma unroll
    for (int g = 0; g < 4; ++g) {
        const int n = g * 16 + wv;
        oih[g] = (n * KB_IH) * FRAG + lane * 16;
        ohh[g] = (n * KB_HH) * FRAG + lane * 16;
    }
    float bias_g[4];
    const float* bias = bias_all + dir * GATES;
#pragma unroll
    for (int g = 0; g < 4; ++g) bias_g[g] = bias[g * HID + wv * 16 + l15];

    for (int i = tid; i < 16 * (HID + 8); i += 512) (&h_lds[0][0])[i] = (__bf16)0.0f;

    v8f c_reg = {};   // cell state: 16 rows x 16 cols per wave, in registers

    for (int s = 0; s < T_SEQ; ++s) {
        const int tg = dir ? (T_SEQ - 1 - s) : s;

        // Defeat LICM without laundering the pointers: redefine the integer
        // offsets each timestep, keeping loads on the global (not flat) path.
#pragma unroll
        for (int g = 0; g < 4; ++g) {
            asm volatile("" : "+v"(oih[g]));
            asm volatile("" : "+v"(ohh[g]));
        }

        // Stage x_t tile (16x128 f32 -> bf16 LDS), coalesced.
        const float* xt = x + ((size_t)tg * BATCH + m0) * IN_DIM;
#pragma unroll
        for (int i = 0; i < 4; ++i) {
            const int idx = tid + i * 512;
            const int rr = idx >> 7, cc = idx & 127;
            xa_lds[rr][cc] = (__bf16)xt[rr * IN_DIM + cc];
        }
        __syncthreads();   // xa ready; prev-step h writes visible

        v8f acc[4];
#pragma unroll
        for (int g = 0; g < 4; ++g)
#pragma unroll
            for (int r = 0; r < 8; ++r) acc[g][r] = bias_g[g];

        v16bf bcur[4], bnxt[4];

        // ---- input projection: gates += x_t * w_ih^T (double-buffered B stream) ----
#pragma unroll
        for (int g = 0; g < 4; ++g) bcur[g] = pack_frag(wbase, oih[g]);
#pragma unroll 2
        for (int kbi = 0; kbi < KB_IH; ++kbi) {
#pragma unroll
            for (int g = 0; g < 4; ++g)
                bnxt[g] = pack_frag(wbase, oih[g] + (kbi + 1) * FRAG); // tail over-read lands in ws
            const v16bf a = lds_afrag(&xa_lds[l15][kbi * 32 + khalf]);
#pragma unroll
            for (int g = 0; g < 4; ++g)
                acc[g] = __builtin_amdgcn_wmma_f32_16x16x32_bf16(
                    false, a, false, bcur[g], (short)0, acc[g], false, false);
#pragma unroll
            for (int g = 0; g < 4; ++g) bcur[g] = bnxt[g];
        }

        // ---- recurrence: gates += h * w_hh^T ----
#pragma unroll
        for (int g = 0; g < 4; ++g) bcur[g] = pack_frag(hbase, ohh[g]);
#pragma unroll 2
        for (int kbi = 0; kbi < KB_HH; ++kbi) {
#pragma unroll
            for (int g = 0; g < 4; ++g)
                bnxt[g] = pack_frag(hbase, ohh[g] + (kbi + 1) * FRAG);
            const v16bf a = lds_afrag(&h_lds[l15][kbi * 32 + khalf]);
#pragma unroll
            for (int g = 0; g < 4; ++g)
                acc[g] = __builtin_amdgcn_wmma_f32_16x16x32_bf16(
                    false, a, false, bcur[g], (short)0, acc[g], false, false);
#pragma unroll
            for (int g = 0; g < 4; ++g) bcur[g] = bnxt[g];
        }

        __syncthreads();   // all h_lds reads complete before overwrite

        // Cell update in registers; write back bf16 h and f32 output.
#pragma unroll
        for (int r = 0; r < 8; ++r) {
            const float iv = fast_sigmoid(acc[0][r]);
            const float fv = fast_sigmoid(acc[1][r]);
            const float gv = fast_tanh(acc[2][r]);
            const float ov = fast_sigmoid(acc[3][r]);
            const float cv = fv * c_reg[r] + iv * gv;
            c_reg[r] = cv;
            const float hv = ov * fast_tanh(cv);
            const int m = mbase + r;
            const int j = wv * 16 + l15;
            h_lds[m][j] = (__bf16)hv;
            out[((size_t)tg * BATCH + (m0 + m)) * (2 * HID) + dir * HID + j] = hv;
        }
    }
}

extern "C" void kernel_launch(void* const* d_in, const int* in_sizes, int n_in,
                              void* d_out, int out_size, void* d_ws, size_t ws_size,
                              hipStream_t stream) {
    (void)in_sizes; (void)n_in; (void)out_size; (void)ws_size;
    const float* x     = (const float*)d_in[0];
    const float* wih_f = (const float*)d_in[1];
    const float* whh_f = (const float*)d_in[2];
    const float* bih_f = (const float*)d_in[3];
    const float* bhh_f = (const float*)d_in[4];
    const float* wih_b = (const float*)d_in[5];
    const float* whh_b = (const float*)d_in[6];
    const float* bih_b = (const float*)d_in[7];
    const float* bhh_b = (const float*)d_in[8];

    // Workspace layout (~1.55 MB, L2-resident):
    //   [0, 512K)          : w_ih fragment pack, both dirs (262144 bf16)
    //   [512K, 1536K)      : w_hh fragment pack, both dirs (524288 bf16)
    //   [1536K, 1536K+8K)  : folded biases f32 (also absorbs 1KB stream over-read)
    char* ws = (char*)d_ws;
    __bf16* wih_pack = (__bf16*)ws;
    __bf16* whh_pack = (__bf16*)(ws + 524288);
    float*  bias     = (float*)(ws + 524288 + 1048576);

    lstm_prep_kernel<<<dim3(512), dim3(256), 0, stream>>>(
        wih_f, whh_f, bih_f, bhh_f, wih_b, whh_b, bih_b, bhh_b,
        wih_pack, whh_pack, bias);

    lstm_scan_kernel<<<dim3(32), dim3(512), 0, stream>>>(
        x, wih_pack, whh_pack, bias, (float*)d_out);
}
